// Pointnet_SA_MSG_9792525435023
// MI455X (gfx1250) — matrix operand order repaired and checked by reading, not compiled, not run on hardware
//
#include <hip/hip_runtime.h>
#include <stdint.h>

// ---------------- problem constants ----------------
constexpr int NB   = 8;      // batches
constexpr int NPTS = 8192;   // input points per batch
constexpr int NPT  = 2048;   // sampled points (npoint)
constexpr float EPS_BN = 0.001f;

typedef __attribute__((ext_vector_type(16))) __bf16 v16bf;
typedef __attribute__((ext_vector_type(8)))  float  v8f;

struct __attribute__((aligned(16))) U4 { uint32_t x, y, z, w; };
struct __attribute__((aligned(16))) Frag32B { U4 a, b; };   // 32 bytes = 16 bf16

__device__ __forceinline__ uint32_t pk_bf16(float a, float b) {
  uint32_t x = __builtin_bit_cast(uint32_t, a);
  uint32_t y = __builtin_bit_cast(uint32_t, b);
  x = (x + 0x7FFFu + ((x >> 16) & 1u)) >> 16;   // round-to-nearest-even
  y = (y + 0x7FFFu + ((y >> 16) & 1u)) >> 16;
  return (x & 0xFFFFu) | (y << 16);
}
__device__ __forceinline__ uint16_t f2bf(float a) {
  uint32_t x = __builtin_bit_cast(uint32_t, a);
  return (uint16_t)((x + 0x7FFFu + ((x >> 16) & 1u)) >> 16);
}

// ---------------- 1) farthest point sampling ----------------
__global__ __launch_bounds__(1024)
void fps_kernel(const float* __restrict__ xyz, int* __restrict__ fps) {
  __shared__ float dist[NPTS];          // 32 KB of the 320 KB WGP LDS
  __shared__ float swv[32];
  __shared__ int   swi[32];
  __shared__ int   sfar;
  const int b = blockIdx.x;
  const float* X = xyz + (size_t)b * NPTS * 3;
  for (int i = threadIdx.x; i < NPTS; i += blockDim.x) dist[i] = 1e10f;
  __syncthreads();
  int far = 0;
  for (int it = 0; it < NPT; ++it) {
    if (threadIdx.x == 0) fps[b * NPT + it] = far;
    const float cx = X[far*3+0], cy = X[far*3+1], cz = X[far*3+2];
    float best = -1.0f; int bi = NPTS;
    for (int i = threadIdx.x; i < NPTS; i += blockDim.x) {
      float dx = X[i*3+0]-cx, dy = X[i*3+1]-cy, dz = X[i*3+2]-cz;
      float d = dx*dx + dy*dy + dz*dz;
      float dd = fminf(dist[i], d);
      dist[i] = dd;
      if (dd > best) { best = dd; bi = i; }     // strict > keeps first occurrence
    }
#pragma unroll
    for (int off = 16; off; off >>= 1) {        // wave32 reduce
      float ob = __shfl_down(best, off);
      int   oi = __shfl_down(bi,   off);
      if (ob > best || (ob == best && oi < bi)) { best = ob; bi = oi; }
    }
    const int wid = threadIdx.x >> 5, lid = threadIdx.x & 31;
    if (lid == 0) { swv[wid] = best; swi[wid] = bi; }
    __syncthreads();
    if (threadIdx.x < 32) {
      best = swv[lid]; bi = swi[lid];
#pragma unroll
      for (int off = 16; off; off >>= 1) {
        float ob = __shfl_down(best, off);
        int   oi = __shfl_down(bi,   off);
        if (ob > best || (ob == best && oi < bi)) { best = ob; bi = oi; }
      }
      if (lid == 0) sfar = bi;
    }
    __syncthreads();
    far = sfar;
    __syncthreads();
  }
}

// ---------------- 2) gather new_xyz into d_out ----------------
__global__ void newxyz_kernel(const float* __restrict__ xyz, const int* __restrict__ fps,
                              float* __restrict__ out) {
  int i = blockIdx.x * blockDim.x + threadIdx.x;   // NB*NPT*3
  if (i >= NB * NPT * 3) return;
  int c = i % 3, bs = i / 3, b = bs / NPT;
  int j = fps[bs];
  out[i] = xyz[((size_t)b * NPTS + j) * 3 + c];
}

// ---------------- 3) ball query: one wave per (b,s), ballot scan ----------------
__global__ void ballquery_kernel(const float* __restrict__ xyz, const float* __restrict__ nxyz,
                                 int* __restrict__ i0, int* __restrict__ i1, int* __restrict__ i2) {
  int gt = blockIdx.x * blockDim.x + threadIdx.x;
  int wave = gt >> 5, lane = gt & 31;
  if (wave >= NB * NPT) return;
  int b = wave / NPT;
  const float* X = xyz + (size_t)b * NPTS * 3;
  const float cx = nxyz[wave*3+0], cy = nxyz[wave*3+1], cz = nxyz[wave*3+2];
  const float r0 = 0.1f*0.1f, r1 = 0.2f*0.2f, r2 = 0.4f*0.4f;
  int c0 = 0, c1 = 0, c2 = 0, f0 = 0, f1 = 0, f2 = 0;
  const unsigned lt = (1u << lane) - 1u;
  for (int base = 0; base < NPTS; base += 32) {
    if (c0 >= 16 && c1 >= 32 && c2 >= 64) break;
    int j = base + lane;                      // NPTS % 32 == 0
    float dx = X[j*3+0]-cx, dy = X[j*3+1]-cy, dz = X[j*3+2]-cz;
    float d2 = dx*dx + dy*dy + dz*dz;
    unsigned m;
    m = __builtin_amdgcn_ballot_w32(d2 < r0);
    if (m && c0 == 0) f0 = base + __ffs(m) - 1;
    if ((m >> lane) & 1u) { int p = c0 + __popc(m & lt); if (p < 16) i0[wave*16 + p] = j; }
    c0 += __popc(m);
    m = __builtin_amdgcn_ballot_w32(d2 < r1);
    if (m && c1 == 0) f1 = base + __ffs(m) - 1;
    if ((m >> lane) & 1u) { int p = c1 + __popc(m & lt); if (p < 32) i1[wave*32 + p] = j; }
    c1 += __popc(m);
    m = __builtin_amdgcn_ballot_w32(d2 < r2);
    if (m && c2 == 0) f2 = base + __ffs(m) - 1;
    if ((m >> lane) & 1u) { int p = c2 + __popc(m & lt); if (p < 64) i2[wave*64 + p] = j; }
    c2 += __popc(m);
  }
  for (int t = c0 + lane; t < 16; t += 32) i0[wave*16 + t] = f0;   // pad with first
  for (int t = c1 + lane; t < 32; t += 32) i1[wave*32 + t] = f1;
  for (int t = c2 + lane; t < 64; t += 32) i2[wave*64 + t] = f2;
}

// ---------------- 4) gather grouped features -> bf16 A buffer (K padded to 32) ----------------
__global__ void gather_kernel(const float* __restrict__ xyz, const float* __restrict__ pts,
                              const float* __restrict__ nxyz, const int* __restrict__ idx,
                              uint32_t* __restrict__ Abuf, int Ks, long Mrows) {
  long r = (long)blockIdx.x * blockDim.x + threadIdx.x;
  if (r >= Mrows) return;
  int  kk = (int)(r % Ks);
  long bs = r / Ks;
  int  b  = (int)(bs / NPT);
  int  j  = idx[bs * Ks + kk];
  const float* P  = pts  + ((size_t)b * NPTS + j) * 6;
  const float* Xj = xyz  + ((size_t)b * NPTS + j) * 3;
  const float* C  = nxyz + bs * 3;
  uint32_t* dst = Abuf + r * 16;                       // 32 bf16 = 16 dwords
  dst[0] = pk_bf16(P[0], P[1]);
  dst[1] = pk_bf16(P[2], P[3]);
  dst[2] = pk_bf16(P[4], P[5]);
  dst[3] = pk_bf16(Xj[0]-C[0], Xj[1]-C[1]);
  dst[4] = pk_bf16(Xj[2]-C[2], 0.f);
#pragma unroll
  for (int i = 5; i < 16; ++i) dst[i] = 0u;            // zero-pad channels 9..31
}

// ---------------- 5) pack weights to bf16 B-fragments ----------------
// Layout per (k_chunk, n_tile): 32 lanes x 8 dwords; lane l, vgpr v, half h:
//   n = l&15, k_local = (l>>4)*16 + 2v + h
__global__ void packw_kernel(const float* __restrict__ W, uint32_t* __restrict__ Wp,
                             int Cin, int Cout, int Kc) {
  int i = blockIdx.x * blockDim.x + threadIdx.x;
  int ntiles = Cout >> 4;
  int total  = (Kc >> 5) * ntiles * 256;
  if (i >= total) return;
  int v = i & 7, lane = (i >> 3) & 31, tile = i >> 8;
  int tn = tile % ntiles, kcI = tile / ntiles;
  int n = tn * 16 + (lane & 15);
  int k = kcI * 32 + ((lane >> 4) << 4) + 2 * v;
  float lo = (k     < Cin) ? W[(long)k       * Cout + n] : 0.f;
  float hi = (k + 1 < Cin) ? W[(long)(k + 1) * Cout + n] : 0.f;
  Wp[i] = pk_bf16(lo, hi);
}

// ---------------- 6) WMMA GEMM: one wave per 16-row m-tile, all NT n-tiles ----------------
// A fragment loaded once per k-chunk, reused across NT WMMAs (A is the HBM-bound operand).
template <int NT>
__global__ __launch_bounds__(256)
void gemm_kernel(const uint32_t* __restrict__ A,   // bf16 rows, Kc elems/row
                 const uint32_t* __restrict__ Wp,  // packed bf16 B-fragments
                 float* __restrict__ G, float* __restrict__ stat,
                 long M, int Kc) {
  constexpr int Cout = NT * 16;
  const int lane = threadIdx.x & 31;
  const long wave = (long)blockIdx.x * (blockDim.x >> 5) + (threadIdx.x >> 5);
  const long mtiles = M >> 4;
  if (wave >= mtiles) return;                     // wave-uniform (EXEC all-1 for WMMA)
  const int  hi  = lane >> 4;
  const long row = wave * 16 + (lane & 15);
  const uint32_t* Arow = A + row * (long)(Kc >> 1);
  v8f acc[NT];
#pragma unroll
  for (int t = 0; t < NT; ++t) acc[t] = (v8f){0.f,0.f,0.f,0.f,0.f,0.f,0.f,0.f};
  for (int kc = 0; kc < Kc; kc += 32) {
    const int kd = kc >> 1;
    Frag32B fa;
    // A k order used here (contiguous, same as packw B): lane<16 -> K[0..15]; lane>=16 -> K[16..31]
    fa.a = *(const U4*)(Arow + kd + hi * 8);
    fa.b = *(const U4*)(Arow + kd + hi * 8 + 4);
    v16bf av = __builtin_bit_cast(v16bf, fa);
    const U4* bbase = (const U4*)Wp + ((long)(kc >> 5) * NT * 32 + lane) * 2;
#pragma unroll
    for (int t = 0; t < NT; ++t) {
      Frag32B fb;
      fb.a = bbase[t * 64 + 0];
      fb.b = bbase[t * 64 + 1];
      v16bf bv = __builtin_bit_cast(v16bf, fb);
      acc[t] = __builtin_amdgcn_wmma_f32_16x16x32_bf16(false, av, false, bv,
                                                       (short)0, acc[t], false, false);
    }
  }
  // C/D layout: vgpr v, lane l -> row = v + 8*(l>=16), col = l&15
  float* gb = G + (wave * 16 + hi * 8) * (long)Cout;
#pragma unroll
  for (int t = 0; t < NT; ++t) {
    const int col = t * 16 + (lane & 15);
    float s = 0.f, q = 0.f;
#pragma unroll
    for (int v = 0; v < 8; ++v) {
      float x = acc[t][v];
      gb[(long)v * Cout + col] = x;
      s += x; q += x * x;
    }
    s += __shfl_xor(s, 16);
    q += __shfl_xor(q, 16);
    if (lane < 16) {
      atomicAdd(&stat[col], s);        // global_atomic_add_f32
      atomicAdd(&stat[128 + col], q);
    }
  }
}

// ---------------- 7) BN + ReLU -> bf16 activations ----------------
__global__ void bnrelu_kernel(const float* __restrict__ G, const float* __restrict__ stat,
                              const float* __restrict__ gamma, const float* __restrict__ beta,
                              uint16_t* __restrict__ Aout, long total, int Cout, float invM) {
  long i = (long)blockIdx.x * blockDim.x + threadIdx.x;
  if (i >= total) return;
  int c = (int)(i % Cout);
  float mean = stat[c] * invM;
  float var  = stat[128 + c] * invM - mean * mean;
  float sc = gamma[c] * rsqrtf(var + EPS_BN);
  float sh = beta[c] - mean * sc;
  Aout[i] = f2bf(fmaxf(G[i] * sc + sh, 0.f));
}

// ---------------- 8) BN + ReLU + max over samples -> d_out ----------------
__global__ void bnmax_kernel(const float* __restrict__ G, const float* __restrict__ stat,
                             const float* __restrict__ gamma, const float* __restrict__ beta,
                             float* __restrict__ out, int Ks, int Cout, int coff, float invM) {
  int i = blockIdx.x * blockDim.x + threadIdx.x;    // NB*NPT*Cout
  if (i >= NB * NPT * Cout) return;
  int c = i % Cout, bs = i / Cout;
  float mean = stat[c] * invM;
  float var  = stat[128 + c] * invM - mean * mean;
  float sc = gamma[c] * rsqrtf(var + EPS_BN);
  float sh = beta[c] - mean * sc;
  const float* g = G + (long)bs * Ks * Cout + c;
  float m = 0.f;                                    // relu >= 0, so 0 is the identity
  for (int kk = 0; kk < Ks; ++kk) m = fmaxf(m, g[(long)kk * Cout] * sc + sh);
  out[(long)bs * 320 + coff + c] = m;
}

__global__ void zero_kernel(float* __restrict__ p, int n) {
  int i = blockIdx.x * blockDim.x + threadIdx.x;
  if (i < n) p[i] = 0.f;
}

// ---------------- host orchestration ----------------
extern "C" void kernel_launch(void* const* d_in, const int* in_sizes, int n_in,
                              void* d_out, int out_size, void* d_ws, size_t ws_size,
                              hipStream_t stream) {
  (void)in_sizes; (void)n_in; (void)out_size; (void)ws_size;
  const float* xyz = (const float*)d_in[0];
  const float* pts = (const float*)d_in[1];
  float* out = (float*)d_out;

  char* w = (char*)d_ws;
  auto alloc = [&](size_t bytes) -> char* {
    char* p = w; w += (bytes + 255) & ~(size_t)255; return p;
  };
  int*   fps   = (int*)  alloc((size_t)NB * NPT * 4);
  int*   idx0  = (int*)  alloc((size_t)NB * NPT * 16 * 4);
  int*   idx1  = (int*)  alloc((size_t)NB * NPT * 32 * 4);
  int*   idx2  = (int*)  alloc((size_t)NB * NPT * 64 * 4);
  float* stats = (float*)alloc(9 * 256 * 4);

  static const int mlp[3][3]  = {{32, 32, 64}, {64, 64, 128}, {64, 96, 128}};
  static const int nsamp[3]   = {16, 32, 64};
  static const int coffs[3]   = {0, 64, 192};

  uint32_t* wp[9];
  int cins[9], kcs[9], couts[9];
  for (int sc = 0; sc < 3; ++sc) {
    int cin = 9;
    for (int l = 0; l < 3; ++l) {
      int li = sc * 3 + l;
      int cout = mlp[sc][l];
      int kc = (cin + 31) & ~31;
      cins[li] = cin; kcs[li] = kc; couts[li] = cout;
      wp[li] = (uint32_t*)alloc((size_t)(kc >> 5) * (cout >> 4) * 256 * 4);
      cin = cout;
    }
  }
  uint16_t* Abuf = (uint16_t*)alloc((size_t)NB * NPT * 64 * 128 * 2); // bf16 activations
  float*    Gbuf = (float*)   alloc((size_t)NB * NPT * 64 * 128 * 4); // fp32 pre-BN

  zero_kernel<<<(9 * 256 + 255) / 256, 256, 0, stream>>>(stats, 9 * 256);
  fps_kernel<<<NB, 1024, 0, stream>>>(xyz, fps);
  newxyz_kernel<<<(NB * NPT * 3 + 255) / 256, 256, 0, stream>>>(xyz, fps, out);
  ballquery_kernel<<<(NB * NPT * 32 + 255) / 256, 256, 0, stream>>>(xyz, out, idx0, idx1, idx2);
  for (int li = 0; li < 9; ++li) {
    const float* W = (const float*)d_in[2 + li * 3 + 0];
    int total = (kcs[li] >> 5) * (couts[li] >> 4) * 256;
    packw_kernel<<<(total + 255) / 256, 256, 0, stream>>>(W, wp[li], cins[li], couts[li], kcs[li]);
  }
  int* idxs[3] = {idx0, idx1, idx2};
  for (int sc = 0; sc < 3; ++sc) {
    long Mrows = (long)NB * NPT * nsamp[sc];
    gather_kernel<<<(int)((Mrows + 255) / 256), 256, 0, stream>>>(
        xyz, pts, out, idxs[sc], (uint32_t*)Abuf, nsamp[sc], Mrows);
    float invM = 1.0f / (float)Mrows;
    for (int l = 0; l < 3; ++l) {
      int li = sc * 3 + l;
      long waves = Mrows >> 4;                 // one wave per 16-row m-tile
      long thr = waves * 32;
      int blocks = (int)((thr + 255) / 256);
      switch (couts[li] >> 4) {
        case 2: gemm_kernel<2><<<blocks, 256, 0, stream>>>(
                    (const uint32_t*)Abuf, wp[li], Gbuf, stats + li * 256, Mrows, kcs[li]);
                break;
        case 4: gemm_kernel<4><<<blocks, 256, 0, stream>>>(
                    (const uint32_t*)Abuf, wp[li], Gbuf, stats + li * 256, Mrows, kcs[li]);
                break;
        case 6: gemm_kernel<6><<<blocks, 256, 0, stream>>>(
                    (const uint32_t*)Abuf, wp[li], Gbuf, stats + li * 256, Mrows, kcs[li]);
                break;
        default: gemm_kernel<8><<<blocks, 256, 0, stream>>>(
                    (const uint32_t*)Abuf, wp[li], Gbuf, stats + li * 256, Mrows, kcs[li]);
                break;
      }
      const float* gamma = (const float*)d_in[2 + li * 3 + 1];
      const float* beta  = (const float*)d_in[2 + li * 3 + 2];
      if (l < 2) {
        long total = Mrows * couts[li];
        bnrelu_kernel<<<(int)((total + 255) / 256), 256, 0, stream>>>(
            Gbuf, stats + li * 256, gamma, beta, Abuf, total, couts[li], invM);
      } else {
        int total = NB * NPT * couts[li];
        bnmax_kernel<<<(total + 255) / 256, 256, 0, stream>>>(
            Gbuf, stats + li * 256, gamma, beta,
            out + (size_t)NB * NPT * 3, nsamp[sc], couts[li], coffs[sc], invM);
      }
    }
  }
}
